// CrossAttention_50251117363518
// MI455X (gfx1250) — matrix operand order repaired and checked by reading, not compiled, not run on hardware
//
#include <hip/hip_runtime.h>
#include <hip/hip_bf16.h>

// ---------------------------------------------------------------------------
// CDNA5 (gfx1250) fused trinity-mask cross-attention.
// bf16 WMMA (v_wmma_f32_16x16x32_bf16) for all four GEMMs + flash attention.
//   - weights pre-converted once to transposed bf16 (contiguous B-operand pairs)
//   - 16x64 output per wave in GEMMs (4 WMMA accumulators, 4x A reuse)
//   - V stored transposed [B,H,d,N] so P@V B-operand pairs are contiguous
//   - global_prefetch for next key block in the flash loop
// ---------------------------------------------------------------------------

typedef __bf16 bf16_t;
typedef bf16_t v16bf __attribute__((ext_vector_type(16)));
typedef float  v8f   __attribute__((ext_vector_type(8)));

#define WMMA_BF16(A, B, C) \
  __builtin_amdgcn_wmma_f32_16x16x32_bf16(false, (A), false, (B), (short)0, (C), false, false)

// reduce across the 16 lanes of a half-wave (rows live per-half in C/D layout)
static __device__ __forceinline__ float half_max(float v) {
#pragma unroll
  for (int off = 1; off <= 8; off <<= 1) v = fmaxf(v, __shfl_xor(v, off, 32));
  return v;
}
static __device__ __forceinline__ float half_sum(float v) {
#pragma unroll
  for (int off = 1; off <= 8; off <<= 1) v += __shfl_xor(v, off, 32);
  return v;
}

// ---------------------------------------------------------------------------
// Weight pre-pass: Wt[col][k] = (bf16)W[k][col]   (transpose + convert, once)
// ---------------------------------------------------------------------------
__global__ void __launch_bounds__(256)
convert_w_kernel(const float* __restrict__ W, bf16_t* __restrict__ Wt,
                 int Kdim, int Ncols)
{
  int idx = blockIdx.x * 256 + threadIdx.x;
  if (idx < Kdim * Ncols) {
    int k = idx / Ncols, col = idx % Ncols;
    Wt[(size_t)col * Kdim + k] = (bf16_t)W[idx];
  }
}

// ---------------------------------------------------------------------------
// GEMM body: each wave owns a 16x64 output strip (4 WMMA tiles).
// A tile (16x32) staged through LDS as bf16, shared by all 8 waves.
// B from pre-transposed bf16 weights: (k,k+1) pairs contiguous.
// ---------------------------------------------------------------------------
template <typename EPILOGUE>
static __device__ __forceinline__ void gemm_16x64(
    const float* __restrict__ X, const bf16_t* __restrict__ Wt,
    int Kdim, bf16_t* Xs, EPILOGUE epi)
{
  const int wave  = threadIdx.x >> 5;
  const int lane  = threadIdx.x & 31;
  const int lrow  = lane & 15;
  const int lhalf = lane >> 4;
  const int row0  = blockIdx.x * 16;
  const int col0  = (blockIdx.y * 8 + wave) * 64;

  v8f acc[4] = {v8f{}, v8f{}, v8f{}, v8f{}};
  for (int k0 = 0; k0 < Kdim; k0 += 32) {
    // cooperative f32 -> bf16 stage of X[row0..+16][k0..+32] (float2 loads)
    {
      int e = threadIdx.x * 2;
      int r = e >> 5, c = e & 31;
      const float2 xv = *(const float2*)(X + (size_t)(row0 + r) * Kdim + k0 + c);
      Xs[e]     = (bf16_t)xv.x;
      Xs[e + 1] = (bf16_t)xv.y;
    }
    __syncthreads();

    // A operand: row = lane%16; K pairs at (v>>2)*16 + half*8 + (v&3)*2
    v16bf a;
#pragma unroll
    for (int v = 0; v < 8; ++v) {
      int kb = ((v >> 2) << 4) + (lhalf << 3) + ((v & 3) << 1);
      a[2 * v]     = Xs[lrow * 32 + kb];
      a[2 * v + 1] = Xs[lrow * 32 + kb + 1];
    }
    // 4 B tiles, 4 accumulating WMMAs (A reused)
#pragma unroll
    for (int dt = 0; dt < 4; ++dt) {
      const bf16_t* wcol = Wt + (size_t)(col0 + dt * 16 + lrow) * Kdim + k0;
      v16bf bm;
#pragma unroll
      for (int v = 0; v < 8; ++v) {
        int kk = ((v >> 2) << 4) + (lhalf << 3) + ((v & 3) << 1);
        bm[2 * v]     = wcol[kk];
        bm[2 * v + 1] = wcol[kk + 1];
      }
      acc[dt] = WMMA_BF16(a, bm, acc[dt]);
    }
    __syncthreads();
  }
  epi(acc, row0, col0, lrow, lhalf);
}

// Q/K/V projection: f32 X @ bf16 Wt + b -> bf16 out.
// MODE 0: head-split [B,H,N,d] (Q,K).  MODE 1: transposed [B,H,d,N] (V).
template <int MODE>
__global__ void __launch_bounds__(256)
proj_gemm_kernel(const float* __restrict__ X, const bf16_t* __restrict__ Wt,
                 const float* __restrict__ bias, bf16_t* __restrict__ out,
                 int Kdim, int Nseq, int numHeads, int headDim, float scale)
{
  __shared__ bf16_t Xs[16 * 32];
  gemm_16x64(X, Wt, Kdim, Xs,
    [&](v8f (&acc)[4], int row0, int col0, int lrow, int lhalf) {
#pragma unroll
      for (int dt = 0; dt < 4; ++dt) {
        const int col = col0 + dt * 16 + lrow;
        const float bv = bias[col];
        const int h  = col / headDim;
        const int dd = col % headDim;
#pragma unroll
        for (int r = 0; r < 8; ++r) {
          int grow = row0 + r + (lhalf << 3);          // row in [0, B*N)
          int bb = grow / Nseq, n = grow % Nseq;
          size_t idx = (MODE == 0)
              ? (((size_t)bb * numHeads + h) * Nseq + n) * headDim + dd
              : (((size_t)bb * numHeads + h) * headDim + dd) * Nseq + n;
          out[idx] = (bf16_t)((acc[dt][r] + bv) * scale);
        }
      }
    });
}

// Output projection: f32 O @ bf16 WoT + bo -> f32 out, plain [M, Ncols].
__global__ void __launch_bounds__(256)
out_gemm_kernel(const float* __restrict__ X, const bf16_t* __restrict__ Wt,
                const float* __restrict__ bias, float* __restrict__ out,
                int Kdim, int Ncols)
{
  __shared__ bf16_t Xs[16 * 32];
  gemm_16x64(X, Wt, Kdim, Xs,
    [&](v8f (&acc)[4], int row0, int col0, int lrow, int lhalf) {
#pragma unroll
      for (int dt = 0; dt < 4; ++dt) {
        const int col = col0 + dt * 16 + lrow;
        const float bv = bias[col];
#pragma unroll
        for (int r = 0; r < 8; ++r) {
          int grow = row0 + r + (lhalf << 3);
          out[(size_t)grow * Ncols + col] = acc[dt][r] + bv;
        }
      }
    });
}

// ---------------------------------------------------------------------------
// Flash attention: one wave per (batch, head, 16-query tile).
// ---------------------------------------------------------------------------
static __device__ __forceinline__ v8f score_tile(
    const bf16_t* __restrict__ Kb, int k16, int headDim,
    v16bf qa0, v16bf qa1, int lrow, int lhalf)
{
  // B operand = K^T: lane (col n = lane%16) reads contiguous d from key row n
  v16bf b0, b1;
  const bf16_t* kr = Kb + (size_t)(k16 + lrow) * headDim;
#pragma unroll
  for (int v = 0; v < 8; ++v) {
    int d0 = ((v >> 2) << 4) + (lhalf << 3) + ((v & 3) << 1);
    b0[2 * v] = kr[d0];       b0[2 * v + 1] = kr[d0 + 1];
    b1[2 * v] = kr[d0 + 32];  b1[2 * v + 1] = kr[d0 + 33];
  }
  v8f s = {};
  s = WMMA_BF16(qa0, b0, s);   // d = 0..31
  s = WMMA_BF16(qa1, b1, s);   // d = 32..63
  return s;
}

__global__ void __launch_bounds__(32)
attn_kernel(const bf16_t* __restrict__ Qh, const bf16_t* __restrict__ Kh,
            const bf16_t* __restrict__ Vt, float* __restrict__ Om,
            const int* __restrict__ imgLenP,
            int N, int numHeads, int headDim)
{
  __shared__ bf16_t Pl[16 * 32];   // P re-layout buffer (C layout -> A layout)

  const int lane  = threadIdx.x & 31;
  const int lrow  = lane & 15;
  const int lhalf = lane >> 4;
  const int qt = blockIdx.x, h = blockIdx.y, b = blockIdx.z;
  const int qi0 = qt * 16;
  const int imgLen = *imgLenP;          // 1536
  const int sit = imgLen / 3;           // 512

  const size_t base = (((size_t)b * numHeads + h) * N) * headDim;
  const bf16_t* Qb  = Qh + base;
  const bf16_t* Kb  = Kh + base;
  const bf16_t* Vtb = Vt + base;        // [d][N] within this (b,h)

  // Preload Q A-operands once (reused for every key block): two 16x32 tiles
  v16bf qa0, qa1;
  {
    const bf16_t* qr = Qb + (size_t)(qi0 + lrow) * headDim;
#pragma unroll
    for (int v = 0; v < 8; ++v) {
      int kb = ((v >> 2) << 4) + (lhalf << 3) + ((v & 3) << 1);
      qa0[2 * v] = qr[kb];      qa0[2 * v + 1] = qr[kb + 1];
      qa1[2 * v] = qr[kb + 32]; qa1[2 * v + 1] = qr[kb + 33];
    }
  }

  // trinity mask -> contiguous key range per query tile
  const bool isText = (qi0 >= imgLen);
  int kstart, kend;
  if (!isText) { int chunk = qi0 / sit; kstart = chunk * sit; kend = kstart + sit; }
  else         { kstart = 0;            kend = qi0 + 16; }   // img + causal text

  float mrun[8], lsum[8];
#pragma unroll
  for (int r = 0; r < 8; ++r) { mrun[r] = -3.0e38f; lsum[r] = 0.0f; }
  v8f o[4] = {v8f{}, v8f{}, v8f{}, v8f{}};

  for (int kb = kstart; kb < kend; kb += 32) {
    // prefetch next key block (K rows + V columns) while computing this one
    if (kb + 32 < kend) {
      __builtin_prefetch(Kb + (size_t)(kb + 32 + lrow) * headDim, 0, 0);
      __builtin_prefetch(Vtb + (size_t)lrow * N + kb + 32, 0, 0);
    }

    // ---- S = Q K^T for two 16-key subtiles (scale pre-folded into Q)
    v8f s0 = score_tile(Kb, kb, headDim, qa0, qa1, lrow, lhalf);
    v8f s1;
    const bool sub1 = (kb + 16) < kend;
    if (sub1) {
      s1 = score_tile(Kb, kb + 16, headDim, qa0, qa1, lrow, lhalf);
    } else {
#pragma unroll
      for (int r = 0; r < 8; ++r) s1[r] = -3.0e38f;
    }
    if (isText) {
#pragma unroll
      for (int r = 0; r < 8; ++r) {
        int i  = qi0 + r + (lhalf << 3);
        int j0 = kb + lrow, j1 = kb + 16 + lrow;
        if (j0 >= imgLen && j0 > i) s0[r] = -3.0e38f;
        if (j1 >= imgLen && j1 > i) s1[r] = -3.0e38f;
      }
    }

    // ---- online softmax (per-lane rows r + 8*half)
    float p0[8], p1[8];
#pragma unroll
    for (int r = 0; r < 8; ++r) {
      float tmax  = half_max(fmaxf(s0[r], s1[r]));
      float mnew  = fmaxf(mrun[r], tmax);
      float alpha = __expf(mrun[r] - mnew);
      mrun[r] = mnew;
      p0[r] = __expf(s0[r] - mnew);
      p1[r] = __expf(s1[r] - mnew);
      lsum[r] = lsum[r] * alpha + half_sum(p0[r] + p1[r]);
      o[0][r] *= alpha; o[1][r] *= alpha; o[2][r] *= alpha; o[3][r] *= alpha;
    }

    // ---- re-layout P (C/D layout) -> A operand via LDS
#pragma unroll
    for (int r = 0; r < 8; ++r) {
      int row = r + (lhalf << 3);
      Pl[row * 32 + lrow]      = (bf16_t)p0[r];
      Pl[row * 32 + 16 + lrow] = (bf16_t)p1[r];
    }
    asm volatile("s_wait_dscnt 0" ::: "memory");   // single-wave block: DS wait suffices
    v16bf pa;
#pragma unroll
    for (int v = 0; v < 8; ++v) {
      int kp = ((v >> 2) << 4) + (lhalf << 3) + ((v & 3) << 1);
      pa[2 * v]     = Pl[lrow * 32 + kp];
      pa[2 * v + 1] = Pl[lrow * 32 + kp + 1];
    }
    asm volatile("s_wait_dscnt 0" ::: "memory");

    // ---- O += P @ V across four 16-wide d tiles (Vt: pairs contiguous in key)
#pragma unroll
    for (int dt = 0; dt < 4; ++dt) {
      const bf16_t* vr = Vtb + (size_t)(dt * 16 + lrow) * N;
      v16bf vb;
#pragma unroll
      for (int v = 0; v < 8; ++v) {
        int j0 = kb + ((v >> 2) << 4) + (lhalf << 3) + ((v & 3) << 1);
        int ja = (j0     < kend) ? j0       : kend - 1;  // clamp: P==0 there
        int jc = (j0 + 1 < kend) ? (j0 + 1) : kend - 1;
        vb[2 * v]     = vr[ja];
        vb[2 * v + 1] = vr[jc];
      }
      o[dt] = WMMA_BF16(pa, vb, o[dt]);
    }
  }

  // ---- normalize + store O in [B, N, C] f32
  const int C = numHeads * headDim;
#pragma unroll
  for (int r = 0; r < 8; ++r) {
    int grow  = qi0 + r + (lhalf << 3);
    float inv = 1.0f / lsum[r];
    size_t rb = ((size_t)b * N + grow) * C + h * headDim;
#pragma unroll
    for (int dt = 0; dt < 4; ++dt)
      Om[rb + dt * 16 + lrow] = o[dt][r] * inv;
  }
}

// ---------------------------------------------------------------------------
// Host launcher
// ---------------------------------------------------------------------------
extern "C" void kernel_launch(void* const* d_in, const int* in_sizes, int n_in,
                              void* d_out, int out_size, void* d_ws, size_t ws_size,
                              hipStream_t stream) {
  // Fixed reference shapes: B=4, N=2048 (512 txt + 1536 img), C=1024, H=16, d=64
  constexpr int B = 4, N = 2048, C = 1024, H = 16, DH = 64;
  constexpr float SCALE = 0.125f;   // 64^-0.5

  const float* q  = (const float*)d_in[0];
  const float* kv = (const float*)d_in[1];
  const float* Wq = (const float*)d_in[2];
  const float* bq = (const float*)d_in[3];
  const float* Wk = (const float*)d_in[4];
  const float* bk = (const float*)d_in[5];
  const float* Wv = (const float*)d_in[6];
  const float* bv = (const float*)d_in[7];
  const float* Wo = (const float*)d_in[8];
  const float* bo = (const float*)d_in[9];
  const int* imgLenP = (const int*)d_in[11];

  // workspace layout:
  //   4x transposed bf16 weights (2 MB each) | Qh | Kh (bf16 [B,H,N,d], 16 MB)
  //   Vt (bf16 [B,H,d,N], 16 MB) | Om (f32 [B,N,C], 32 MB)
  const size_t wElems    = (size_t)C * C;
  const size_t headElems = (size_t)B * H * N * DH;
  bf16_t* WqT = (bf16_t*)d_ws;
  bf16_t* WkT = WqT + wElems;
  bf16_t* WvT = WkT + wElems;
  bf16_t* WoT = WvT + wElems;
  bf16_t* Qh  = WoT + wElems;
  bf16_t* Kh  = Qh + headElems;
  bf16_t* Vt  = Kh + headElems;
  float*  Om  = (float*)(Vt + headElems);

  const int cvtBlocks = (int)(wElems / 256);
  convert_w_kernel<<<cvtBlocks, 256, 0, stream>>>(Wq, WqT, C, C);
  convert_w_kernel<<<cvtBlocks, 256, 0, stream>>>(Wk, WkT, C, C);
  convert_w_kernel<<<cvtBlocks, 256, 0, stream>>>(Wv, WvT, C, C);
  convert_w_kernel<<<cvtBlocks, 256, 0, stream>>>(Wo, WoT, C, C);

  const dim3 gGemm((B * N) / 16, (C / 64) / 8);   // (512, 2), 8 waves x 16x64
  proj_gemm_kernel<0><<<gGemm, 256, 0, stream>>>(q,  WqT, bq, Qh, C, N, H, DH, SCALE);
  proj_gemm_kernel<0><<<gGemm, 256, 0, stream>>>(kv, WkT, bk, Kh, C, N, H, DH, 1.0f);
  proj_gemm_kernel<1><<<gGemm, 256, 0, stream>>>(kv, WvT, bv, Vt, C, N, H, DH, 1.0f);

  attn_kernel<<<dim3(N / 16, H, B), 32, 0, stream>>>(Qh, Kh, Vt, Om, imgLenP, N, H, DH);

  out_gemm_kernel<<<gGemm, 256, 0, stream>>>(Om, WoT, bo, (float*)d_out, C, C);
}
